// UltrametricCausalSelfAttention_24292335026387
// MI455X (gfx1250) — matrix operand order repaired and checked by reading, not compiled, not run on hardware
//
#include <hip/hip_runtime.h>

// CDNA5 / gfx1250, wave32. All matrix-like math on v_wmma_f32_16x16x32_f16:
// QKV projections, digit projections (q@Wdq / k@Wdk), w@V, output projection.
// - LDS tiles in WMMA-fragment order => fragment reads are b128 loads.
// - A tiles staged with global_load_async_to_lds_b128 (ASYNCcnt path).
// - Double-buffered LDS: one barrier per K-step, staging overlaps WMMA.

typedef __attribute__((ext_vector_type(16))) _Float16 v16h;
typedef __attribute__((ext_vector_type(8)))  _Float16 v8h;
typedef __attribute__((ext_vector_type(8)))  float    v8f;

#define B_  2
#define T_  1024
#define C_  1024
#define H_  16
#define HD_ 64
#define KD_ 8
#define BT_ (B_*T_)

union U16h { v16h v; v8h h[2]; };

__device__ __forceinline__ _Float16 lo_h(unsigned d) {
    return __builtin_bit_cast(_Float16, (unsigned short)(d & 0xffffu));
}
__device__ __forceinline__ _Float16 hi_h(unsigned d) {
    return __builtin_bit_cast(_Float16, (unsigned short)(d >> 16));
}
__device__ __forceinline__ void load8(float* d, const float* p) {
    float4 a = *(const float4*)p, b = *(const float4*)(p + 4);
    d[0]=a.x; d[1]=a.y; d[2]=a.z; d[3]=a.w;
    d[4]=b.x; d[5]=b.y; d[6]=b.z; d[7]=b.w;
}

// fragment-order swizzle: element (n, k) of a (BK=32 x BN) f16 tile.
// B-fragment of tile n16 = 32 contiguous bytes at ((n16*2+kHi)<<8)+(nLo<<4).
__device__ __forceinline__ int fswz(int n, int k) {
    return (((n >> 4) * 2 + (k >> 4)) << 8) + ((n & 15) << 4) + (k & 15);
}
__device__ __forceinline__ void swz_store8(_Float16* buf, int n, int k, uint4 d) {
    buf[fswz(n + 0, k)] = lo_h(d.x); buf[fswz(n + 1, k)] = hi_h(d.x);
    buf[fswz(n + 2, k)] = lo_h(d.y); buf[fswz(n + 3, k)] = hi_h(d.y);
    buf[fswz(n + 4, k)] = lo_h(d.z); buf[fswz(n + 5, k)] = hi_h(d.z);
    buf[fswz(n + 6, k)] = lo_h(d.w); buf[fswz(n + 7, k)] = hi_h(d.w);
}

#define ASYNC_LDS_B128(ldsaddr, gaddr)                                     \
    asm volatile("global_load_async_to_lds_b128 %0, %1, off"               \
                 :: "v"(ldsaddr), "v"(gaddr) : "memory")
#define WAIT_ASYNC0() asm volatile("s_wait_asynccnt 0x0" ::: "memory")

// ---------------------------------------------------------------------------
// Stage 1: f32 -> f16, 8 elements/thread
// ---------------------------------------------------------------------------
__global__ void cvt_f32_to_f16_kernel(const float* __restrict__ in,
                                      _Float16* __restrict__ out, int n8) {
    int i = blockIdx.x * blockDim.x + threadIdx.x;
    int stride = gridDim.x * blockDim.x;
    for (; i < n8; i += stride) {
        const float4* p = (const float4*)in + (size_t)i * 2;
        float4 a = p[0], b = p[1];
        v8h o;
        o[0] = (_Float16)a.x; o[1] = (_Float16)a.y;
        o[2] = (_Float16)a.z; o[3] = (_Float16)a.w;
        o[4] = (_Float16)b.x; o[5] = (_Float16)b.y;
        o[6] = (_Float16)b.z; o[7] = (_Float16)b.w;
        *(v8h*)(out + (size_t)i * 8) = o;
    }
}

// ---------------------------------------------------------------------------
// Stage 2/5: f16 GEMM, f32 accum.  C[M,N] = A[M,K] * B[K,N]
// 256 threads = 8 waves; 64(M) x 128(N) block tile, BK=32; wave = 32x32.
// Double-buffered: async-A + staged-B for step s+1 issued before compute(s).
// Fragment maps as used here (wave32; A and B share contiguous K order):
//   A elem e (lane L): K = e + 16*(L>>4), M = L&15
//   B elem e (lane L): K = e + 16*(L>>4),                N = L&15
//   C/D row r (lane L): M = r + 8*(L>>4),                N = L&15
// ---------------------------------------------------------------------------
__global__ void __launch_bounds__(256)
gemm_f16_wmma_kernel(const _Float16* __restrict__ A,
                     const _Float16* __restrict__ Bm,
                     float* __restrict__ Cm, int M, int N, int K) {
    __shared__ alignas(16) _Float16 As[2][64][40];   // row-major, 80B stride
    __shared__ alignas(32) _Float16 Bs[2][4096];     // fragment-order swizzle
    const int tid    = threadIdx.x;
    const int lane   = tid & 31, wave = tid >> 5;
    const int laneHi = lane >> 4, laneLo = lane & 15;
    const int wm = wave & 1, wn = wave >> 1;          // 2 x 4 wave grid
    const int m0 = blockIdx.x * 64, n0 = blockIdx.y * 128;

    const int arow = tid >> 2, achk = (tid & 3) * 8;
    const unsigned alds[2] = {
        (unsigned)(unsigned long long)&As[0][arow][achk],
        (unsigned)(unsigned long long)&As[1][arow][achk]};
    const _Float16* gA = A + (size_t)(m0 + arow) * K + achk;

    const int brow0 = tid >> 4, brow1 = brow0 + 16, bcol = (tid & 15) * 8;
    const _Float16* gB0 = Bm + (size_t)brow0 * N + n0 + bcol;
    const _Float16* gB1 = Bm + (size_t)brow1 * N + n0 + bcol;
    const size_t strideB = (size_t)32 * N;

    ASYNC_LDS_B128(alds[0], (unsigned long long)gA);
    {
        uint4 d0 = *(const uint4*)gB0;
        uint4 d1 = *(const uint4*)gB1;
        swz_store8(Bs[0], bcol, brow0, d0);
        swz_store8(Bs[0], bcol, brow1, d1);
    }
    gA += 32; gB0 += strideB; gB1 += strideB;

    const int nsteps = K >> 5;
    v8f acc[2][2] = {};
    for (int step = 0; step < nsteps; ++step) {
        const int cur = step & 1;
        WAIT_ASYNC0();
        __syncthreads();
        if (step + 1 < nsteps) {
            ASYNC_LDS_B128(alds[cur ^ 1], (unsigned long long)gA);
            uint4 d0 = *(const uint4*)gB0;
            uint4 d1 = *(const uint4*)gB1;
            swz_store8(Bs[cur ^ 1], bcol, brow0, d0);
            swz_store8(Bs[cur ^ 1], bcol, brow1, d1);
            __builtin_prefetch(gB0 + strideB, 0, 1);
            gA += 32; gB0 += strideB; gB1 += strideB;
        }
        U16h a[2], b[2];
        #pragma unroll
        for (int mi = 0; mi < 2; ++mi) {
            const _Float16* arp = &As[cur][wm * 32 + mi * 16 + laneLo][0];
            a[mi].h[0] = *(const v8h*)(arp + laneHi * 16);
            a[mi].h[1] = *(const v8h*)(arp + laneHi * 16 + 8);
        }
        #pragma unroll
        for (int ni = 0; ni < 2; ++ni) {
            int n16 = wn * 2 + ni;
            b[ni].v = *(const v16h*)&Bs[cur][((n16 * 2 + laneHi) << 8) + (laneLo << 4)];
        }
        #pragma unroll
        for (int mi = 0; mi < 2; ++mi)
            #pragma unroll
            for (int ni = 0; ni < 2; ++ni)
                acc[mi][ni] = __builtin_amdgcn_wmma_f32_16x16x32_f16(
                    false, a[mi].v, false, b[ni].v, (short)0, acc[mi][ni],
                    false, false);
    }
    #pragma unroll
    for (int mi = 0; mi < 2; ++mi)
        #pragma unroll
        for (int ni = 0; ni < 2; ++ni)
            #pragma unroll
            for (int r = 0; r < 8; ++r) {
                int m = m0 + wm * 32 + mi * 16 + r + (laneHi << 3);
                int n = n0 + wn * 32 + ni * 16 + laneLo;
                Cm[(size_t)m * N + n] = acc[mi][ni][r];
            }
}

// ---------------------------------------------------------------------------
// Stage 3: RoPE + RMS + digit projection on WMMA + V cast.
// One wave = 16 time-rows of one (b,h).  Lanes (L, L+16) split a row's 64
// dims into the A-fragment K-chunks (disjoint halves; RMS via shfl_xor 16).
// raw = norm(rope(q)) @ Wd  ->  2 chained WMMAs (Wd zero-padded N=8->16).
// Digit byte of row r = bits[7:0] of __ballot(raw>0); row r+8 = bits[23:16].
// ---------------------------------------------------------------------------
__device__ __forceinline__ void digit_half(
        const float* __restrict__ src, const float* __restrict__ Wd,
        unsigned char* __restrict__ dst, size_t rowbase, size_t dstbase,
        const float* c0, const float* c1, const float* s0, const float* s1,
        int lane, int laneHi, int laneLo) {
    float i0[8], i1[8], i2[8], i3[8];
    const float* p = src + rowbase + 8 * laneHi;
    load8(i0, p); load8(i1, p + 16); load8(i2, p + 32); load8(i3, p + 48);

    float o0[8], o1[8], o2[8], o3[8];
    float ss = 0.f;
    #pragma unroll
    for (int c = 0; c < 8; ++c) {
        o0[c] =  i0[c] * c0[c] + i2[c] * s0[c];     // dims  0..31 (rope lo)
        o1[c] =  i1[c] * c1[c] + i3[c] * s1[c];
        o2[c] = -i0[c] * s0[c] + i2[c] * c0[c];     // dims 32..63 (rope hi)
        o3[c] = -i1[c] * s1[c] + i3[c] * c1[c];
        ss += o0[c]*o0[c] + o1[c]*o1[c] + o2[c]*o2[c] + o3[c]*o3[c];
    }
    float tot = ss + __shfl_xor(ss, 16, 32);        // full-row sum of squares
    float rn = rsqrtf(tot / HD_ + 1e-6f);

    v16h a1, a2;                                    // K 0..31 and K 32..63
    #pragma unroll
    for (int c = 0; c < 8; ++c) {
        a1[c]     = (_Float16)(o0[c] * rn);
        a1[c + 8] = (_Float16)(o1[c] * rn);
        a2[c]     = (_Float16)(o2[c] * rn);
        a2[c + 8] = (_Float16)(o3[c] * rn);
    }
    v16h b1, b2;                                    // Wd as B-fragments
    #pragma unroll
    for (int e = 0; e < 16; ++e) {
        int k1 = (e & 7) + 16 * (e >> 3) + 8 * laneHi;
        b1[e] = (_Float16)((laneLo < KD_) ? Wd[k1 * KD_ + laneLo] : 0.f);
        b2[e] = (_Float16)((laneLo < KD_) ? Wd[(k1 + 32) * KD_ + laneLo] : 0.f);
    }
    v8f raw = {};
    raw = __builtin_amdgcn_wmma_f32_16x16x32_f16(false, a1, false, b1,
                                                 (short)0, raw, false, false);
    raw = __builtin_amdgcn_wmma_f32_16x16x32_f16(false, a2, false, b2,
                                                 (short)0, raw, false, false);
    #pragma unroll
    for (int r = 0; r < 8; ++r) {
        unsigned long long bm = __ballot(raw[r] > 0.f);
        if (lane == 0) {
            dst[dstbase + r]     = (unsigned char)(bm & 0xff);
            dst[dstbase + r + 8] = (unsigned char)((bm >> 16) & 0xff);
        }
    }
}

__global__ void __launch_bounds__(256)
rope_digit_wmma_kernel(const float* __restrict__ qf,
                       const float* __restrict__ kf,
                       const float* __restrict__ vf,
                       const float* __restrict__ cosT,
                       const float* __restrict__ sinT,
                       const float* __restrict__ Wdq,
                       const float* __restrict__ Wdk,
                       unsigned char* __restrict__ dqb,
                       unsigned char* __restrict__ dkb,
                       _Float16* __restrict__ vh) {
    const int tid = threadIdx.x, lane = tid & 31, wave = tid >> 5;
    const int laneHi = lane >> 4, laneLo = lane & 15;
    const int g  = blockIdx.x * 8 + wave;           // tile = 16 rows of a head
    const int bh = g >> 6, tt = g & 63;             // T/16 = 64 tiles per head
    const int t0 = tt * 16, t = t0 + laneLo;
    const int b = bh / H_, h = bh % H_;
    const size_t rowbase = (size_t)(b * T_ + t) * C_ + h * HD_;
    const size_t dstbase = (size_t)bh * T_ + t0;

    float c0[8], c1[8], s0[8], s1[8];               // this lane's rope angles
    {
        const float* cp = cosT + t * (HD_ / 2) + 8 * laneHi;
        const float* sp = sinT + t * (HD_ / 2) + 8 * laneHi;
        load8(c0, cp); load8(c1, cp + 16);
        load8(s0, sp); load8(s1, sp + 16);
    }
    digit_half(qf, Wdq, dqb, rowbase, dstbase, c0, c1, s0, s1,
               lane, laneHi, laneLo);
    digit_half(kf, Wdk, dkb, rowbase, dstbase, c0, c1, s0, s1,
               lane, laneHi, laneLo);

    // V cast/transpose: this lane's 32 dims of row t -> (b,h,t,hd) f16
    const float* vp = vf + rowbase + 8 * laneHi;
    _Float16* vo = vh + ((size_t)bh * T_ + t) * HD_ + 8 * laneHi;
    #pragma unroll
    for (int cch = 0; cch < 4; ++cch) {
        float4 x0 = *(const float4*)(vp + cch * 16);
        float4 x1 = *(const float4*)(vp + cch * 16 + 4);
        v8h o;
        o[0]=(_Float16)x0.x; o[1]=(_Float16)x0.y; o[2]=(_Float16)x0.z; o[3]=(_Float16)x0.w;
        o[4]=(_Float16)x1.x; o[5]=(_Float16)x1.y; o[6]=(_Float16)x1.z; o[7]=(_Float16)x1.w;
        *(v8h*)(vo + cch * 16) = o;
    }
}

// ---------------------------------------------------------------------------
// Stage 4: ultrametric causal attention.
// w(q,k) = 2^min(ctz(dq^dk),8) if k<=q else 0, built in-register into the f16
// A-fragment (exact in f16).  num = w @ V, 4 WMMAs per 32-key tile.
// Double-buffered V tile (regs prefetched one tile ahead, one barrier/iter);
// all dk bytes of the causal prefix staged in LDS once, read as b64.
// ---------------------------------------------------------------------------
__global__ void __launch_bounds__(256)
ultra_attn_wmma_kernel(const unsigned char* __restrict__ dqb,
                       const unsigned char* __restrict__ dkb,
                       const _Float16* __restrict__ vh,
                       _Float16* __restrict__ yh) {
    __shared__ alignas(32) _Float16 Vs[2][2048];  // swizzled 32x64 V tiles
    __shared__ unsigned Ks32[256];                // up to 1024 dk bytes
    __shared__ float Dens[128];
    const int tid    = threadIdx.x;
    const int lane   = tid & 31, wave = tid >> 5;
    const int laneHi = lane >> 4, laneLo = lane & 15;
    const int bh = blockIdx.y, b = bh / H_, h = bh % H_;
    const int qbase = blockIdx.x * 128;
    const int qm = qbase + wave * 16 + laneLo;    // this lane's query row
    const unsigned myq = dqb[(size_t)bh * T_ + qm];
    const int nkt = (qbase + 128) >> 5;           // causal tile bound

    if (tid < nkt * 8)
        Ks32[tid] = ((const unsigned*)(dkb + (size_t)bh * T_))[tid];

    const int vrow = tid >> 3, vcol = (tid & 7) * 8;
    const _Float16* gV = vh + ((size_t)bh * T_ + vrow) * HD_ + vcol;
    uint4 vreg = *(const uint4*)gV;               // prefetch tile 0

    v8f acc[4] = {};
    float den = 0.f;
    for (int kt = 0; kt < nkt; ++kt) {
        const int cur = kt & 1;
        swz_store8(Vs[cur], vcol, vrow, vreg);    // publish prefetched tile
        __syncthreads();
        if (kt + 1 < nkt) {
            gV += (size_t)32 * HD_;
            vreg = *(const uint4*)gV;
        }
        const int kb = kt << 5;
        const unsigned char* kbytes = (const unsigned char*)Ks32;
        uint2 g0 = *(const uint2*)(kbytes + kb + 16 * laneHi);
        uint2 g1 = *(const uint2*)(kbytes + kb + 16 * laneHi + 8);
        const unsigned kw[4] = {g0.x, g0.y, g1.x, g1.y};

        v16h aw;
        #pragma unroll
        for (int e = 0; e < 16; ++e) {
            int klocal = e + (laneHi << 4);
            int kk = kb + klocal;
            unsigned dkbyte = (kw[e >> 2] >> ((e & 3) * 8)) & 0xffu;
            float w = 0.f;
            if (kk <= qm) {
                int lcp = __builtin_ctz((myq ^ dkbyte) | 0x100u);
                w = (float)(1u << lcp);           // ALPHA=2 -> exact in f16
            }
            aw[e] = (_Float16)w;
            den += w;
        }
        #pragma unroll
        for (int nt = 0; nt < 4; ++nt) {
            v16h bv = *(const v16h*)&Vs[cur][((nt * 2 + laneHi) << 8) + (laneLo << 4)];
            acc[nt] = __builtin_amdgcn_wmma_f32_16x16x32_f16(
                false, aw, false, bv, (short)0, acc[nt], false, false);
        }
    }
    float dfull = den + __shfl_xor(den, 16, 32);
    __syncthreads();
    if (lane < 16) Dens[wave * 16 + lane] = dfull;
    __syncthreads();

    #pragma unroll
    for (int r = 0; r < 8; ++r) {
        int m = r + (laneHi << 3);
        int q = qbase + wave * 16 + m;
        float d = fmaxf(Dens[wave * 16 + m], 1e-9f);
        size_t orow = (size_t)(b * T_ + q) * C_ + h * HD_;
        #pragma unroll
        for (int nt = 0; nt < 4; ++nt)
            yh[orow + nt * 16 + laneLo] = (_Float16)(acc[nt][r] / d);
    }
}

// ---------------------------------------------------------------------------
// Host-side orchestration
// ---------------------------------------------------------------------------
extern "C" void kernel_launch(void* const* d_in, const int* in_sizes, int n_in,
                              void* d_out, int out_size, void* d_ws, size_t ws_size,
                              hipStream_t stream) {
    (void)in_sizes; (void)n_in; (void)out_size; (void)ws_size;
    const float* x     = (const float*)d_in[0];
    const float* cosT  = (const float*)d_in[1];
    const float* sinT  = (const float*)d_in[2];
    const float* Wq    = (const float*)d_in[3];
    const float* Wk    = (const float*)d_in[4];
    const float* Wv    = (const float*)d_in[5];
    const float* Wproj = (const float*)d_in[6];
    const float* Wdq   = (const float*)d_in[7];
    const float* Wdk   = (const float*)d_in[8];
    float* out = (float*)d_out;

    char* ws = (char*)d_ws;
    size_t off = 0;
    auto carve = [&](size_t bytes) {
        off = (off + 255) & ~(size_t)255;
        void* p = ws + off;
        off += bytes;
        return p;
    };
    _Float16* xh  = (_Float16*)carve((size_t)BT_ * C_ * 2);
    _Float16* Wqh = (_Float16*)carve((size_t)C_ * C_ * 2);
    _Float16* Wkh = (_Float16*)carve((size_t)C_ * C_ * 2);
    _Float16* Wvh = (_Float16*)carve((size_t)C_ * C_ * 2);
    _Float16* Wph = (_Float16*)carve((size_t)C_ * C_ * 2);
    float*    qf  = (float*)   carve((size_t)BT_ * C_ * 4);
    float*    kf  = (float*)   carve((size_t)BT_ * C_ * 4);
    float*    vf  = (float*)   carve((size_t)BT_ * C_ * 4);
    _Float16* vhh = (_Float16*)carve((size_t)B_ * H_ * T_ * HD_ * 2);
    unsigned char* dqb = (unsigned char*)carve((size_t)B_ * H_ * T_);
    unsigned char* dkb = (unsigned char*)carve((size_t)B_ * H_ * T_);
    _Float16* yh  = (_Float16*)carve((size_t)BT_ * C_ * 2);

    cvt_f32_to_f16_kernel<<<256, 256, 0, stream>>>(x,     xh,  BT_ * C_ / 8);
    cvt_f32_to_f16_kernel<<<256, 256, 0, stream>>>(Wq,    Wqh, C_ * C_ / 8);
    cvt_f32_to_f16_kernel<<<256, 256, 0, stream>>>(Wk,    Wkh, C_ * C_ / 8);
    cvt_f32_to_f16_kernel<<<256, 256, 0, stream>>>(Wv,    Wvh, C_ * C_ / 8);
    cvt_f32_to_f16_kernel<<<256, 256, 0, stream>>>(Wproj, Wph, C_ * C_ / 8);

    dim3 ggrid(BT_ / 64, C_ / 128);
    gemm_f16_wmma_kernel<<<ggrid, 256, 0, stream>>>(xh, Wqh, qf, BT_, C_, C_);
    gemm_f16_wmma_kernel<<<ggrid, 256, 0, stream>>>(xh, Wkh, kf, BT_, C_, C_);
    gemm_f16_wmma_kernel<<<ggrid, 256, 0, stream>>>(xh, Wvh, vf, BT_, C_, C_);

    // B*H*(T/16) wave-tiles / 8 waves per block = 256 blocks
    rope_digit_wmma_kernel<<<256, 256, 0, stream>>>(
        qf, kf, vf, cosT, sinT, Wdq, Wdk, dqb, dkb, vhh);

    dim3 agrid(T_ / 128, B_ * H_);
    ultra_attn_wmma_kernel<<<agrid, 256, 0, stream>>>(dqb, dkb, vhh, yh);

    gemm_f16_wmma_kernel<<<ggrid, 256, 0, stream>>>(yh, Wph, out, BT_, C_, C_);
}
